// MultiHeadAttention_6691559047629
// MI455X (gfx1250) — compile-verified, matrix operand order checked by reading, not run on hardware
//
#include <hip/hip_runtime.h>

typedef _Float16 half_t;
typedef __attribute__((ext_vector_type(16))) _Float16 v16h;
typedef __attribute__((ext_vector_type(8)))  _Float16 v8h;
typedef __attribute__((ext_vector_type(8)))  float    v8f;
typedef __attribute__((ext_vector_type(4)))  float    v4f;

#define WMMA_F16(a, b, c) \
  __builtin_amdgcn_wmma_f32_16x16x32_f16(false, (a), false, (b), (short)0, (c), false, false)

static __device__ __forceinline__ v16h cat8(v8h lo, v8h hi) {
  return __builtin_shufflevector(lo, hi, 0,1,2,3,4,5,6,7,8,9,10,11,12,13,14,15);
}

// ---- CDNA5 async global->LDS copies (ASYNCcnt path), guarded so the file
// ---- still compiles if the toolchain doesn't declare the builtins.
#if __has_builtin(__builtin_amdgcn_global_load_async_to_lds_b128)
#define HAVE_ASYNC_LDS 1
typedef int av4i __attribute__((vector_size(16)));                 // gcc-style int4
typedef __attribute__((address_space(1))) av4i av4i_g;             // global int4
typedef __attribute__((address_space(3))) av4i av4i_l;             // LDS int4
static __device__ __forceinline__ void async_copy16(const void* g, void* l) {
  av4i_g* gp = (av4i_g*)(unsigned long long)(uintptr_t)g;
  av4i_l* lp = (av4i_l*)(unsigned int)(uintptr_t)l;   // low 32 bits = LDS offset
  __builtin_amdgcn_global_load_async_to_lds_b128(gp, lp, 0, 0);
}
static __device__ __forceinline__ void async_wait0() {
#if __has_builtin(__builtin_amdgcn_s_wait_asynccnt)
  __builtin_amdgcn_s_wait_asynccnt(0);
#else
  asm volatile("s_wait_asynccnt 0" ::: "memory");
#endif
}
#endif

// ---------------------------------------------------------------------------
// Tiled GEMM with bias:  C[M,N] = A[M,K] @ W[K,N] + bias[N]
// Block tile 64x64, K-step 32. 256 threads = 8 waves; each wave computes one
// 16-row x 32-col strip as two v_wmma_f32_16x16x32_f16 accumulators.
// A is f32 (projections) or f16 (attention output); C is f16 or f32.
// ---------------------------------------------------------------------------
template<bool A_IS_F32, bool OUT_F32>
__global__ __launch_bounds__(256) void gemm_bias_kernel(
    const void* __restrict__ Ap, const float* __restrict__ W,
    const float* __restrict__ bias, void* __restrict__ Cp,
    int M, int N, int K)
{
  constexpr int BK = 32;
  __shared__ half_t As[64 * BK];   // [m][k]
  __shared__ half_t Bs[64 * BK];   // [n][k]  (W tile transposed)

  const int t    = threadIdx.x;
  const int wave = t >> 5;
  const int lane = t & 31;
  const int bm   = blockIdx.y * 64;
  const int bn   = blockIdx.x * 64;

  const int wm = (wave & 3) * 16;   // wave's row subtile
  const int wn = (wave >> 2) * 32;  // wave's col strip (2 subtiles)

  // staging-load indices
  const int arow = t >> 2;          // 0..63
  const int acol = (t & 3) * 8;     // 0..24
  const int wrow = t >> 3;          // k local 0..31
  const int wcol = (t & 7) * 8;     // n local 0..56

  const int r15  = lane & 15;
  const int koff = (lane >> 4) * 8;   // f16 A/B fragment K-offset per half-wave
  const int mofs = (lane >> 4) * 8;   // f32 C fragment M-offset per half-wave

  v8f acc0 = {}; v8f acc1 = {};

  for (int k0 = 0; k0 < K; k0 += BK) {
    // ---- stage A tile (convert to f16 if needed) ----
    if (A_IS_F32) {
      const float* A = (const float*)Ap;
      const float* src = A + (size_t)(bm + arow) * K + k0 + acol;
      v4f a0 = *(const v4f*)src;
      v4f a1 = *(const v4f*)(src + 4);
      half_t* dst = &As[arow * BK + acol];
      dst[0]=(half_t)a0[0]; dst[1]=(half_t)a0[1]; dst[2]=(half_t)a0[2]; dst[3]=(half_t)a0[3];
      dst[4]=(half_t)a1[0]; dst[5]=(half_t)a1[1]; dst[6]=(half_t)a1[2]; dst[7]=(half_t)a1[3];
    } else {
      const half_t* A = (const half_t*)Ap;
      const half_t* src = A + (size_t)(bm + arow) * K + k0 + acol;
      half_t* dst = &As[arow * BK + acol];
#ifdef HAVE_ASYNC_LDS
      async_copy16(src, dst);       // f16 tile: straight global->LDS DMA
#else
      *(v8h*)dst = *(const v8h*)src;
#endif
    }
    // ---- stage W tile, transposed to [n][k] ----
    {
      const float* src = W + (size_t)(k0 + wrow) * N + bn + wcol;
      v4f w0 = *(const v4f*)src;
      v4f w1 = *(const v4f*)(src + 4);
      Bs[(wcol+0)*BK + wrow] = (half_t)w0[0];
      Bs[(wcol+1)*BK + wrow] = (half_t)w0[1];
      Bs[(wcol+2)*BK + wrow] = (half_t)w0[2];
      Bs[(wcol+3)*BK + wrow] = (half_t)w0[3];
      Bs[(wcol+4)*BK + wrow] = (half_t)w1[0];
      Bs[(wcol+5)*BK + wrow] = (half_t)w1[1];
      Bs[(wcol+6)*BK + wrow] = (half_t)w1[2];
      Bs[(wcol+7)*BK + wrow] = (half_t)w1[3];
    }
    // ---- prefetch next K-step tiles into GL2 while this step computes ----
    if (k0 + BK < K) {
      if (A_IS_F32) {
        const float* A = (const float*)Ap;
        __builtin_prefetch(A + (size_t)(bm + arow) * K + k0 + BK + acol, 0, 1);
      } else {
        const half_t* A = (const half_t*)Ap;
        __builtin_prefetch(A + (size_t)(bm + arow) * K + k0 + BK + acol, 0, 1);
      }
      __builtin_prefetch(W + (size_t)(k0 + BK + wrow) * N + bn + wcol, 0, 1);
    }
#if defined(HAVE_ASYNC_LDS)
    if (!A_IS_F32) async_wait0();
#endif
    __syncthreads();

    // ---- fragments + WMMA ----
    v8h alo = *(const v8h*)&As[(wm + r15) * BK + koff];
    v8h ahi = *(const v8h*)&As[(wm + r15) * BK + koff + 16];
    v16h afrag = cat8(alo, ahi);

    v8h b0lo = *(const v8h*)&Bs[(wn + r15) * BK + koff];
    v8h b0hi = *(const v8h*)&Bs[(wn + r15) * BK + koff + 16];
    v16h bfrag0 = cat8(b0lo, b0hi);
    v8h b1lo = *(const v8h*)&Bs[(wn + 16 + r15) * BK + koff];
    v8h b1hi = *(const v8h*)&Bs[(wn + 16 + r15) * BK + koff + 16];
    v16h bfrag1 = cat8(b1lo, b1hi);

    acc0 = WMMA_F16(afrag, bfrag0, acc0);
    acc1 = WMMA_F16(afrag, bfrag1, acc1);
    __syncthreads();
  }

  // ---- epilogue: bias + store ----
#pragma unroll
  for (int r = 0; r < 8; ++r) {
    int m  = bm + wm + mofs + r;
    int n0 = bn + wn + r15;
    int n1 = n0 + 16;
    float v0 = acc0[r] + bias[n0];
    float v1 = acc1[r] + bias[n1];
    if (OUT_F32) {
      float* C = (float*)Cp;
      C[(size_t)m * N + n0] = v0;
      C[(size_t)m * N + n1] = v1;
    } else {
      half_t* C = (half_t*)Cp;
      C[(size_t)m * N + n0] = (half_t)v0;
      C[(size_t)m * N + n1] = (half_t)v1;
    }
  }
}

// ---------------------------------------------------------------------------
// Flash attention (causal), per (b, h, 64-row q tile). DH = 64.
// QK^T and P@V both via v_wmma_f32_16x16x32_f16; online softmax in LDS.
// Q/K tiles staged with GLOBAL_LOAD_ASYNC_TO_LDS_B128 when available.
// ---------------------------------------------------------------------------
__global__ __launch_bounds__(256) void flash_attn_kernel(
    const half_t* __restrict__ Qh, const half_t* __restrict__ Kh,
    const half_t* __restrict__ Vh, half_t* __restrict__ Oh)
{
  constexpr int S = 2048, D = 1024;
  __shared__ half_t Qt[64 * 64];   // [q][d]
  __shared__ half_t Kt[64 * 64];   // [kv][d]   (== B-fragment layout for QK^T)
  __shared__ half_t Vt[64 * 64];   // [d][kv]   (transposed V, B-fragment for PV)
  __shared__ half_t Pt[64 * 64];   // probabilities, f16
  __shared__ float  Sb[64 * 64];   // raw scores, f32
  __shared__ float  mrun[64], lrun[64], resc[64];

  const int t    = threadIdx.x;
  const int wave = t >> 5;
  const int lane = t & 31;
  const int qb   = blockIdx.x * 64;
  const int h    = blockIdx.y;
  const int b    = blockIdx.z;
  const size_t rowbase = (size_t)b * S * D;
  const int colbase = h * 64;

  const int li  = t >> 2;          // row 0..63 for staging
  const int ld0 = (t & 3) * 16;    // col 0..48

  // load Q tile once (async DMA into LDS when available)
  {
    const half_t* src = Qh + rowbase + (size_t)(qb + li) * D + colbase + ld0;
    half_t* dst = &Qt[li * 64 + ld0];
#ifdef HAVE_ASYNC_LDS
    async_copy16(src, dst);
    async_copy16(src + 8, dst + 8);
#else
    *(v8h*)dst       = *(const v8h*)src;
    *(v8h*)(dst + 8) = *(const v8h*)(src + 8);
#endif
  }
  if (t < 64) { mrun[t] = -1e30f; lrun[t] = 0.0f; }

  const int wm   = (wave & 3) * 16;
  const int wn   = (wave >> 2) * 32;
  const int r15  = lane & 15;
  const int koff = (lane >> 4) * 8;
  const int mofs = (lane >> 4) * 8;

  v8f o0 = {}; v8f o1 = {};

  for (int kb = 0; kb <= qb; kb += 64) {
    __syncthreads();  // previous iteration done with Kt/Vt/Pt/Sb/resc

    // ---- stage K tile [kv][d] (async) and V tile transposed [d][kv] ----
    {
      const half_t* src = Kh + rowbase + (size_t)(kb + li) * D + colbase + ld0;
      half_t* dst = &Kt[li * 64 + ld0];
#ifdef HAVE_ASYNC_LDS
      async_copy16(src, dst);
      async_copy16(src + 8, dst + 8);
#else
      *(v8h*)dst       = *(const v8h*)src;
      *(v8h*)(dst + 8) = *(const v8h*)(src + 8);
#endif
    }
    {
      const half_t* src = Vh + rowbase + (size_t)(kb + li) * D + colbase + ld0;
      v8h v0 = *(const v8h*)src;
      v8h v1 = *(const v8h*)(src + 8);
#pragma unroll
      for (int j = 0; j < 8; ++j) Vt[(ld0 + j) * 64 + li]     = v0[j];
#pragma unroll
      for (int j = 0; j < 8; ++j) Vt[(ld0 + 8 + j) * 64 + li] = v1[j];
    }
#ifdef HAVE_ASYNC_LDS
    async_wait0();   // Q tile (iter 0) + K tile landed in LDS
#endif
    __syncthreads();

    // ---- S = Q K^T over d (two K-steps of 32) ----
    v8f s0 = {}; v8f s1 = {};
#pragma unroll
    for (int ks = 0; ks < 2; ++ks) {
      const int kk = ks * 32;
      v16h a  = cat8(*(const v8h*)&Qt[(wm + r15) * 64 + kk + koff],
                     *(const v8h*)&Qt[(wm + r15) * 64 + kk + koff + 16]);
      v16h b0 = cat8(*(const v8h*)&Kt[(wn + r15) * 64 + kk + koff],
                     *(const v8h*)&Kt[(wn + r15) * 64 + kk + koff + 16]);
      v16h b1 = cat8(*(const v8h*)&Kt[(wn + 16 + r15) * 64 + kk + koff],
                     *(const v8h*)&Kt[(wn + 16 + r15) * 64 + kk + koff + 16]);
      s0 = WMMA_F16(a, b0, s0);
      s1 = WMMA_F16(a, b1, s1);
    }

    // ---- scale + causal mask, spill to LDS f32 ----
#pragma unroll
    for (int r = 0; r < 8; ++r) {
      const int ml = wm + mofs + r;
      const int qg = qb + ml;
      const int n0 = wn + r15, n1 = n0 + 16;
      float x0 = s0[r] * 0.125f; if (kb + n0 > qg) x0 = -1e30f;
      float x1 = s1[r] * 0.125f; if (kb + n1 > qg) x1 = -1e30f;
      Sb[ml * 64 + n0] = x0;
      Sb[ml * 64 + n1] = x1;
    }
    __syncthreads();

    // ---- online softmax: 4 threads per row, each scans the full row ----
    {
      const int row = t >> 2;
      const int seg = (t & 3) * 16;
      float vmax = -1e30f;
#pragma unroll
      for (int j = 0; j < 64; ++j) vmax = fmaxf(vmax, Sb[row * 64 + j]);
      const float mold = mrun[row];
      const float mnew = fmaxf(mold, vmax);
      float sum = 0.0f;
#pragma unroll
      for (int j = 0; j < 64; ++j) sum += __expf(Sb[row * 64 + j] - mnew);
#pragma unroll
      for (int j = 0; j < 16; ++j)
        Pt[row * 64 + seg + j] = (half_t)__expf(Sb[row * 64 + seg + j] - mnew);
      if ((t & 3) == 0) {
        const float f = __expf(mold - mnew);
        resc[row] = f;
        mrun[row] = mnew;
        lrun[row] = lrun[row] * f + sum;
      }
    }
    __syncthreads();

    // ---- rescale accumulators, then O += P @ V ----
#pragma unroll
    for (int r = 0; r < 8; ++r) {
      const float f = resc[wm + mofs + r];
      o0[r] *= f; o1[r] *= f;
    }
#pragma unroll
    for (int ks = 0; ks < 2; ++ks) {
      const int kk = ks * 32;
      v16h a  = cat8(*(const v8h*)&Pt[(wm + r15) * 64 + kk + koff],
                     *(const v8h*)&Pt[(wm + r15) * 64 + kk + koff + 16]);
      v16h b0 = cat8(*(const v8h*)&Vt[(wn + r15) * 64 + kk + koff],
                     *(const v8h*)&Vt[(wn + r15) * 64 + kk + koff + 16]);
      v16h b1 = cat8(*(const v8h*)&Vt[(wn + 16 + r15) * 64 + kk + koff],
                     *(const v8h*)&Vt[(wn + 16 + r15) * 64 + kk + koff + 16]);
      o0 = WMMA_F16(a, b0, o0);
      o1 = WMMA_F16(a, b1, o1);
    }
  }

  // ---- normalize and store f16 ----
#pragma unroll
  for (int r = 0; r < 8; ++r) {
    const int ml  = wm + mofs + r;
    const float inv = 1.0f / lrun[ml];
    const int n0 = wn + r15, n1 = n0 + 16;
    half_t* dst = Oh + rowbase + (size_t)(qb + ml) * D + colbase;
    dst[n0] = (half_t)(o0[r] * inv);
    dst[n1] = (half_t)(o1[r] * inv);
  }
}

// ---------------------------------------------------------------------------
extern "C" void kernel_launch(void* const* d_in, const int* in_sizes, int n_in,
                              void* d_out, int out_size, void* d_ws, size_t ws_size,
                              hipStream_t stream) {
  (void)in_sizes; (void)n_in; (void)out_size; (void)ws_size;
  constexpr int B = 4, S = 2048, D = 1024, H = 16;
  constexpr int M = B * S;                 // 8192
  constexpr size_t PLANE = (size_t)M * D;  // elements per f16 plane

  const float* q  = (const float*)d_in[0];
  const float* k  = (const float*)d_in[1];
  const float* v  = (const float*)d_in[2];
  // d_in[3] = mask: causal, implemented analytically
  const float* wq = (const float*)d_in[4];
  const float* bq = (const float*)d_in[5];
  const float* wk = (const float*)d_in[6];
  const float* bk = (const float*)d_in[7];
  const float* wv = (const float*)d_in[8];
  const float* bv = (const float*)d_in[9];
  const float* wo = (const float*)d_in[10];
  const float* bo = (const float*)d_in[11];

  half_t* qh = (half_t*)d_ws;
  half_t* kh = qh + PLANE;
  half_t* vh = kh + PLANE;
  half_t* ah = vh + PLANE;

  dim3 gg(D / 64, M / 64);     // 16 x 128 blocks
  dim3 blk(256);
  gemm_bias_kernel<true, false><<<gg, blk, 0, stream>>>(q, wq, bq, qh, M, D, D);
  gemm_bias_kernel<true, false><<<gg, blk, 0, stream>>>(k, wk, bk, kh, M, D, D);
  gemm_bias_kernel<true, false><<<gg, blk, 0, stream>>>(v, wv, bv, vh, M, D, D);

  dim3 ga(S / 64, H, B);       // 32 x 16 x 4 blocks
  flash_attn_kernel<<<ga, blk, 0, stream>>>(qh, kh, vh, ah);

  gemm_bias_kernel<false, true><<<gg, blk, 0, stream>>>(ah, wo, bo, (float*)d_out, M, D, D);
}
